// PVMLayer16_6511170421521
// MI455X (gfx1250) — matrix operand order repaired and checked by reading, not compiled, not run on hardware
//
#include <hip/hip_runtime.h>
#include <hip/hip_bf16.h>
#include <math.h>

typedef __attribute__((ext_vector_type(16))) _Float16 v16h;
typedef __attribute__((ext_vector_type(8)))  float    v8f;

#define NB 8
#define CCH 1024
#define LL 2304
#define NCHUNK 16
#define DM 64
#define DI 128
#define DST 16
#define DTR 4
#define NCB (NB*NCHUNK)            // 128 chunk-batches
#define ROWS ((size_t)NCB*LL)      // 294912 mamba rows
#define BL  ((size_t)NB*LL)        // 18432 outer rows

#if defined(__gfx1250__) && __has_builtin(__builtin_amdgcn_global_load_async_to_lds_b32)
#define HAS_ASYNC_LDS 1
#else
#define HAS_ASYNC_LDS 0
#endif

// ---------------- WMMA fragment helpers (gfx1250 wave32 layouts) ----------------

// A fragment 16x32 f16 from row-major f32 (lda = K stride)
__device__ __forceinline__ v16h frag_a_f32(const float* A, int lda, long row0, int k0, int lane) {
  int m = lane & 15;
  int koff = k0 + ((lane >> 4) << 3);   // +0 or +8
  const float* p = A + (size_t)(row0 + m) * lda + koff;
  v16h f;
#pragma unroll
  for (int i = 0; i < 8; ++i) { f[i] = (_Float16)p[i]; f[i + 8] = (_Float16)p[i + 16]; }
  return f;
}

// A fragment from row-major f16
__device__ __forceinline__ v16h frag_a_f16(const _Float16* A, int lda, long row0, int k0, int lane) {
  int m = lane & 15;
  int koff = k0 + ((lane >> 4) << 3);
  const _Float16* p = A + (size_t)(row0 + m) * lda + koff;
  v16h f;
#pragma unroll
  for (int i = 0; i < 8; ++i) { f[i] = p[i]; f[i + 8] = p[i + 16]; }
  return f;
}

// A fragment from LDS slab [16 rows][40 halfs] (padded stride -> conflict-free)
__device__ __forceinline__ v16h frag_a_lds(const _Float16* sl, int lane) {
  int m = lane & 15;
  int klo = (lane >> 4) << 3;
  const _Float16* p = sl + m * 40 + klo;
  v16h f;
#pragma unroll
  for (int i = 0; i < 8; ++i) { f[i] = p[i]; f[i + 8] = p[i + 16]; }
  return f;
}

// B fragment 32x16 from pre-converted f16 weights, row-major (N rows, ldk = K stride)
__device__ __forceinline__ v16h frag_b_f16(const _Float16* W, int ldk, int n0, int k0, int lane) {
  int n = lane & 15;
  int koff = k0 + ((lane >> 4) << 3);   // +0 or +8
  const _Float16* p = W + (size_t)(n0 + n) * ldk + koff;
  v16h f;
#pragma unroll
  for (int i = 0; i < 8; ++i) { f[i] = p[i]; f[i + 8] = p[i + 16]; }
  return f;
}

__device__ __forceinline__ v8f wmma_f16(v16h a, v16h b, v8f c) {
  return __builtin_amdgcn_wmma_f32_16x16x32_f16(false, a, false, b, (short)0, c, false, false);
}

__device__ __forceinline__ void wait_async_lds() {
#if HAS_ASYNC_LDS
#if __has_builtin(__builtin_amdgcn_s_wait_asynccnt)
  __builtin_amdgcn_s_wait_asynccnt(0);
#else
  asm volatile("s_wait_asynccnt 0x0" ::: "memory");
#endif
#endif
}

// ---------------- Kernel 0: one-shot weight f32 -> f16 conversion ----------------
__global__ void cvt_f16(const float* __restrict__ src, _Float16* __restrict__ dst, int n) {
  int i = blockIdx.x * 256 + threadIdx.x;
  if (i < n) dst[i] = (_Float16)src[i];
}

// ---------------- Kernel 1: LN over C + chunk-scatter to xc (f32) ----------------
__global__ void ln1_kernel(const float* __restrict__ x, const float* __restrict__ g,
                           const float* __restrict__ be, float* __restrict__ xc) {
  int b  = blockIdx.x / (LL / 32);
  int l0 = (blockIdx.x % (LL / 32)) * 32;
  int tx = threadIdx.x, ty = threadIdx.y;
  const float* xb = x + (size_t)b * CCH * LL;
  float s = 0.f, sq = 0.f;
#pragma unroll 4
  for (int j = 0; j < 128; ++j) {
    int c = ty + 8 * j;
    float v = xb[(size_t)c * LL + l0 + tx];
    s += v; sq += v * v;
  }
  __shared__ float ssum[8][32], ssq[8][32], smu[32], srs[32];
  ssum[ty][tx] = s; ssq[ty][tx] = sq;
  __syncthreads();
  if (ty == 0) {
    float S = 0.f, Q = 0.f;
#pragma unroll
    for (int p = 0; p < 8; ++p) { S += ssum[p][tx]; Q += ssq[p][tx]; }
    float mu = S * (1.f / CCH);
    float var = Q * (1.f / CCH) - mu * mu;
    smu[tx] = mu; srs[tx] = rsqrtf(var + 1e-5f);
  }
  __syncthreads();
  float mu = smu[tx], rs = srs[tx];
#pragma unroll 4
  for (int j = 0; j < 128; ++j) {
    int c = ty + 8 * j;
    float v = (xb[(size_t)c * LL + l0 + tx] - mu) * rs * g[c] + be[c];
    size_t cb = (size_t)b * NCHUNK + (c >> 6);
    xc[(cb * LL + l0 + tx) * DM + (c & 63)] = v;
  }
}

// ---------------- Kernel 2: in_proj GEMM (ROWS x 64) @ f16 W (256 x 64), 16x64 wave tile ----------------
__global__ void inproj_gemm(const float* __restrict__ xc, const _Float16* __restrict__ Wh,
                            float* __restrict__ xm, float* __restrict__ z) {
  int wave = (int)((blockIdx.x * blockDim.x + threadIdx.x) >> 5);
  int lane = threadIdx.x & 31;
  int  ntq = wave & 3;            // 4 groups of 64 output cols
  long mt  = wave >> 2;
  long row0 = mt * 16;
  int nb = ntq * 64;
  v8f acc[4] = {};
#pragma unroll
  for (int k0 = 0; k0 < DM; k0 += 32) {
    v16h a = frag_a_f32(xc, DM, row0, k0, lane);
#pragma unroll
    for (int j = 0; j < 4; ++j) {
      v16h b = frag_b_f16(Wh, DM, nb + j * 16, k0, lane);
      acc[j] = wmma_f16(a, b, acc[j]);
    }
  }
  // wave-uniform destination select: cols [0,128) -> xm, [128,256) -> z
  float* dst  = (ntq < 2) ? xm : z;
  int colbase = (ntq < 2) ? ntq * 64 : (ntq - 2) * 64;
  int n = lane & 15, mh = (lane >> 4) << 3;
#pragma unroll
  for (int j = 0; j < 4; ++j)
#pragma unroll
    for (int i = 0; i < 8; ++i)
      dst[(size_t)(row0 + mh + i) * DI + colbase + j * 16 + n] = acc[j][i];
}

// ---------------- Kernel 3: depthwise causal conv(4) + SiLU ----------------
__global__ void conv_silu(const float* __restrict__ xm, const float* __restrict__ cw,
                          const float* __restrict__ cb_, float* __restrict__ xconv) {
  size_t idx = (size_t)blockIdx.x * blockDim.x + threadIdx.x;   // ROWS*DI
  int d = idx & (DI - 1);
  size_t r = idx >> 7;
  int l = (int)(r % LL);
  float acc = cb_[d];
#pragma unroll
  for (int k = 0; k < 4; ++k) {
    int lk = l + k - 3;
    if (lk >= 0) acc += cw[d * 4 + k] * xm[(r + (k - 3)) * DI + d];
  }
  xconv[idx] = acc / (1.f + __expf(-acc));
}

// ---------------- Kernel 4: x_proj (N=36) + delta = softplus(dt_raw @ dt_w^T + b) ----------------
__global__ void xproj_dt(const float* __restrict__ xconv, const float* __restrict__ xpw,
                         const float* __restrict__ dtw, const float* __restrict__ dtb,
                         float* __restrict__ xdbl, float* __restrict__ delta) {
  size_t r = blockIdx.x;
  int t = threadIdx.x;
  __shared__ float srow[DI];
  __shared__ float sdt[DTR];
  srow[t] = xconv[r * DI + t];
  __syncthreads();
  if (t < DTR + 2 * DST) {
    const float* w = xpw + (size_t)t * DI;
    float acc = 0.f;
#pragma unroll 8
    for (int k = 0; k < DI; ++k) acc += w[k] * srow[k];
    xdbl[r * 36 + t] = acc;
    if (t < DTR) sdt[t] = acc;
  }
  __syncthreads();
  float acc = dtb[t];
#pragma unroll
  for (int j = 0; j < DTR; ++j) acc += sdt[j] * dtw[t * DTR + j];
  delta[r * DI + t] = (acc > 20.f) ? acc : log1pf(__expf(acc));
}

// ---------------- Kernel 5: selective scan, one lane per (channel, state) ----------------
__global__ void scan_kernel(const float* __restrict__ delta, const float* __restrict__ xdbl,
                            const float* __restrict__ xconv, const float* __restrict__ z,
                            const float* __restrict__ A_log, const float* __restrict__ D_p,
                            _Float16* __restrict__ y16) {
  int t  = threadIdx.x;
  int s  = t & 15;                       // state index
  int dl = t >> 4;                       // channel-in-block (0..15)
  int cb = blockIdx.x >> 3;              // chunk-batch
  int d  = (blockIdx.x & 7) * 16 + dl;   // channel
  float a  = -__expf(A_log[d * DST + s]);
  float Dp = D_p[d];
  float h  = 0.f;
  for (int l = 0; l < LL; ++l) {
    size_t r = (size_t)cb * LL + l;
    float dt = delta[r * DI + d];        // broadcast across 16 lanes
    float u  = xconv[r * DI + d];
    float zv = z[r * DI + d];
    float Bv = xdbl[r * 36 + DTR + s];
    float Cv = xdbl[r * 36 + DTR + DST + s];
    h = __expf(dt * a) * h + dt * Bv * u;
    float yc = h * Cv;
    yc += __shfl_xor(yc, 1, 32);
    yc += __shfl_xor(yc, 2, 32);
    yc += __shfl_xor(yc, 4, 32);
    yc += __shfl_xor(yc, 8, 32);
    if (s == 0) {
      float y = (yc + u * Dp) * (zv / (1.f + __expf(-zv)));
      y16[r * DI + d] = (_Float16)y;
    }
  }
}

// ---------------- Kernel 6: out_proj GEMM (K=128, 16x64 wave tile) + skip -> (B,L,C) ----------------
__global__ void outproj_gemm(const _Float16* __restrict__ y16, const _Float16* __restrict__ Wh,
                             const float* __restrict__ xc, const float* __restrict__ skip_scale,
                             float* __restrict__ ym) {
  long wave = (long)((blockIdx.x * blockDim.x + threadIdx.x) >> 5);
  int lane = threadIdx.x & 31;
  long row0 = wave * 16;                 // one wave covers all 64 output cols
  v8f acc[4] = {};
#pragma unroll
  for (int k0 = 0; k0 < DI; k0 += 32) {
    v16h a = frag_a_f16(y16, DI, row0, k0, lane);
#pragma unroll
    for (int j = 0; j < 4; ++j) {
      v16h b = frag_b_f16(Wh, DI, j * 16, k0, lane);
      acc[j] = wmma_f16(a, b, acc[j]);
    }
  }
  float ss = skip_scale[0];
  int n = lane & 15, mh = (lane >> 4) << 3;
#pragma unroll
  for (int i = 0; i < 8; ++i) {
    size_t r = row0 + mh + i;
    int cb = (int)(r / LL), l = (int)(r % LL);
    int bb = cb >> 4, nc = cb & 15;
    size_t obase = ((size_t)bb * LL + l) * CCH + nc * DM;
#pragma unroll
    for (int j = 0; j < 4; ++j) {
      int col = j * 16 + n;
      ym[obase + col] = acc[j][i] + ss * xc[r * DM + col];
    }
  }
}

// ---------------- Kernel 7: LN over C, emit f16 rows for final GEMM ----------------
__global__ void ln2_kernel(const float* __restrict__ ym, const float* __restrict__ g,
                           const float* __restrict__ be, _Float16* __restrict__ out16) {
  size_t row = blockIdx.x;
  int t = threadIdx.x;
  const float* p = ym + row * CCH;
  float s = 0.f, sq = 0.f;
#pragma unroll
  for (int j = t; j < CCH; j += 256) { float v = p[j]; s += v; sq += v * v; }
  __shared__ float rs_[256], rq_[256];
  rs_[t] = s; rq_[t] = sq;
  __syncthreads();
  for (int off = 128; off > 0; off >>= 1) {
    if (t < off) { rs_[t] += rs_[t + off]; rq_[t] += rq_[t + off]; }
    __syncthreads();
  }
  float mu = rs_[0] * (1.f / CCH);
  float var = rq_[0] * (1.f / CCH) - mu * mu;
  float rstd = rsqrtf(var + 1e-5f);
#pragma unroll
  for (int j = t; j < CCH; j += 256) {
    float v = (p[j] - mu) * rstd * g[j] + be[j];
    out16[row * CCH + j] = (_Float16)v;
  }
}

// ---------------- Kernel 8: final GEMM (BL x 1024) @ f16 W + bias -> (B,O,L) ----------------
// A-slab (16 rows x 32 halfs, block-uniform) staged via async global->LDS copy, double-buffered.
__global__ void final_gemm(const _Float16* __restrict__ A, const _Float16* __restrict__ Wh,
                           const float* __restrict__ bias, float* __restrict__ out) {
  long wave = (long)((blockIdx.x * blockDim.x + threadIdx.x) >> 5);
  int lane = threadIdx.x & 31;
  int w    = (threadIdx.x >> 5) & 7;
  int  ng  = (int)(wave & 15);           // 16 groups of 64 output cols
  long mt  = wave >> 4;                  // block-uniform (8 waves = 8 consecutive ng)
  long row0 = mt * 16;
  int n0 = ng * 64;

  __shared__ _Float16 sA[2][16][40];     // padded: stride 20 dwords -> conflict-free
  int t     = threadIdx.x;
  int arow  = t >> 4;                    // 0..15
  int acol  = (t & 15) * 2;              // half index, step 2 (4 bytes per thread)

  // stage slab k0 into buffer buf (1 KB, one b32 per thread)
  auto stage = [&](int k0, int buf) {
    const _Float16* src = A + (size_t)(row0 + arow) * CCH + k0 + acol;
    _Float16* dst = &sA[buf][arow][acol];
#if HAS_ASYNC_LDS
    __builtin_amdgcn_global_load_async_to_lds_b32(
        (__attribute__((address_space(1))) int*)src,
        (__attribute__((address_space(3))) int*)dst, 0, 0);
#else
    *(int*)dst = *(const int*)src;
#endif
  };

  v8f acc[4] = {};
  stage(0, 0);
  wait_async_lds();
  __syncthreads();
  for (int kb = 0; kb < CCH / 32; ++kb) {
    int k0 = kb * 32;
    int buf = kb & 1;
    if (kb + 1 < CCH / 32) stage(k0 + 32, buf ^ 1);
    v16h a = frag_a_lds(&sA[buf][0][0], lane);
#pragma unroll
    for (int j = 0; j < 4; ++j) {
      v16h b = frag_b_f16(Wh, CCH, n0 + j * 16, k0, lane);
      acc[j] = wmma_f16(a, b, acc[j]);
    }
    wait_async_lds();
    __syncthreads();
  }

  // stage 64(o) x 16(l) tile in LDS (stride-20 padding), store coalesced float4
  __shared__ float sT[8][64][20];
  int n = lane & 15, mh = (lane >> 4) << 3;
#pragma unroll
  for (int j = 0; j < 4; ++j) {
    int ol = j * 16 + n;
    float bv = bias[n0 + ol];
#pragma unroll
    for (int i = 0; i < 8; ++i) sT[w][ol][mh + i] = acc[j][i] + bv;
  }
  int bb = (int)(row0 / LL), l0 = (int)(row0 % LL);
#pragma unroll
  for (int q = 0; q < 8; ++q) {
    int id = lane + 32 * q;
    int ol = id >> 2, quad = (id & 3) * 4;
    float4 v = make_float4(sT[w][ol][quad], sT[w][ol][quad + 1],
                           sT[w][ol][quad + 2], sT[w][ol][quad + 3]);
    *(float4*)(out + ((size_t)bb * CCH + n0 + ol) * LL + l0 + quad) = v;
  }
}

// ---------------- Host launcher ----------------
extern "C" void kernel_launch(void* const* d_in, const int* in_sizes, int n_in,
                              void* d_out, int out_size, void* d_ws, size_t ws_size,
                              hipStream_t stream) {
  const float* x       = (const float*)d_in[0];
  const float* ln_g    = (const float*)d_in[1];
  const float* ln_b    = (const float*)d_in[2];
  const float* skip_s  = (const float*)d_in[3];
  const float* in_w    = (const float*)d_in[4];
  const float* conv_w  = (const float*)d_in[5];
  const float* conv_b  = (const float*)d_in[6];
  const float* xproj_w = (const float*)d_in[7];
  const float* dt_w    = (const float*)d_in[8];
  const float* dt_b    = (const float*)d_in[9];
  const float* A_log   = (const float*)d_in[10];
  const float* D_p     = (const float*)d_in[11];
  const float* out_w   = (const float*)d_in[12];
  const float* proj_w  = (const float*)d_in[13];
  const float* proj_b  = (const float*)d_in[14];
  float* out = (float*)d_out;

  char* ws = (char*)d_ws;
  const size_t SZ_XC    = ROWS * DM * sizeof(float);       // 75.5 MB
  const size_t SZ_ROWDI = ROWS * DI * sizeof(float);       // 151 MB
  const size_t SZ_XDBL  = ROWS * 36 * sizeof(float);       // 42.5 MB
  const size_t SZ_Y16   = ROWS * DI * sizeof(_Float16);    // 75.5 MB

  float*     xcF    = (float*)(ws);
  float*     xm     = (float*)(ws + SZ_XC);
  float*     zbuf   = (float*)(ws + SZ_XC + SZ_ROWDI);
  float*     xconv  = (float*)(ws + SZ_XC + 2 * SZ_ROWDI);
  float*     xdbl   = (float*)(ws + SZ_XC + 3 * SZ_ROWDI);
  _Float16*  y16    = (_Float16*)(ws + SZ_XC + 3 * SZ_ROWDI + SZ_XDBL);
  char*      wbase  = ws + SZ_XC + 3 * SZ_ROWDI + SZ_XDBL + SZ_Y16;
  _Float16*  in_wh  = (_Float16*)(wbase);
  _Float16*  out_wh = (_Float16*)(wbase + 2 * 256 * 64 * sizeof(_Float16));
  _Float16*  proj_wh= (_Float16*)(wbase + 2 * (256 * 64 + 64 * 128) * sizeof(_Float16));
  float*     delta  = xm;                 // xm dead after conv
  float*     ym     = zbuf;               // z dead after scan
  _Float16*  ln2o   = (_Float16*)xconv;   // xconv dead after scan
  (void)ws_size; (void)in_sizes; (void)n_in; (void)out_size;

  // 0) one-shot weight conversions to f16
  cvt_f16<<<dim3((256 * 64 + 255) / 256), dim3(256), 0, stream>>>(in_w, in_wh, 256 * 64);
  cvt_f16<<<dim3((64 * 128 + 255) / 256), dim3(256), 0, stream>>>(out_w, out_wh, 64 * 128);
  cvt_f16<<<dim3((1024 * 1024 + 255) / 256), dim3(256), 0, stream>>>(proj_w, proj_wh, 1024 * 1024);
  // 1) LN + chunk scatter
  ln1_kernel<<<dim3(NB * (LL / 32)), dim3(32, 8), 0, stream>>>(x, ln_g, ln_b, xcF);
  // 2) in_proj GEMM: 16x64 wave tiles
  inproj_gemm<<<dim3((unsigned)((ROWS / 16) * 4 / 8)), dim3(256), 0, stream>>>(xcF, in_wh, xm, zbuf);
  // 3) conv + silu
  conv_silu<<<dim3((unsigned)(ROWS * DI / 256)), dim3(256), 0, stream>>>(xm, conv_w, conv_b, xconv);
  // 4) x_proj + delta
  xproj_dt<<<dim3((unsigned)ROWS), dim3(DI), 0, stream>>>(xconv, xproj_w, dt_w, dt_b, xdbl, delta);
  // 5) selective scan (lane-per-state, shuffle reduce)
  scan_kernel<<<dim3(NCB * 8), dim3(256), 0, stream>>>(delta, xdbl, xconv, zbuf, A_log, D_p, y16);
  // 6) out_proj + skip
  outproj_gemm<<<dim3((unsigned)((ROWS / 16) / 8)), dim3(256), 0, stream>>>(y16, out_wh, xcF, skip_s, ym);
  // 7) LN2 -> f16
  ln2_kernel<<<dim3((unsigned)BL), dim3(256), 0, stream>>>(ym, ln_g, ln_b, ln2o);
  // 8) final projection GEMM -> d_out (B, O, L)
  final_gemm<<<dim3((unsigned)((BL / 16) * 16 / 8)), dim3(256), 0, stream>>>(ln2o, proj_wh, proj_b, out);
}
